// CausalSelfAttention_69011534512786
// MI455X (gfx1250) — matrix operand order repaired and checked by reading, not compiled, not run on hardware
//
#include <hip/hip_runtime.h>
#include <hip/hip_bf16.h>
#include <stdint.h>

// ---------------------------------------------------------------------------
// Types / fragment helpers for V_WMMA_F32_16X16X32_BF16 (wave32)
// ---------------------------------------------------------------------------
typedef __attribute__((ext_vector_type(16))) __bf16 v16bf;
typedef __attribute__((ext_vector_type(8)))  float  v8f;

union Frag16 { uint4 u[2]; v16bf v; };

#define WMMA_BF16(a, b, c) \
  __builtin_amdgcn_wmma_f32_16x16x32_bf16(false, (a), false, (b), (short)0, (c), false, false)

__device__ __forceinline__ unsigned short f2bf(float x) {
  unsigned int u = __float_as_uint(x);
  unsigned int r = u + 0x7FFFu + ((u >> 16) & 1u);   // round-to-nearest-even
  return (unsigned short)(r >> 16);
}

// ---- DPP 16-lane-row reductions (pure VALU, no LDS pipe) -------------------
// dpp_ctrl: 0xB1 quad_perm[1,0,3,2] (xor1), 0x4E quad_perm[2,3,0,1] (xor2),
//           0x141 row_half_mirror, 0x140 row_mirror. Rows of 16 lanes reduce
//           independently => lanes 0-15 and 16-31 each get their own result.
template <int CTRL>
__device__ __forceinline__ float dpp_xf(float x) {
  int r = __builtin_amdgcn_update_dpp(0, __float_as_int(x), CTRL, 0xf, 0xf, true);
  return __int_as_float(r);
}
__device__ __forceinline__ float red16_max(float x) {
  x = fmaxf(x, dpp_xf<0xB1>(x));
  x = fmaxf(x, dpp_xf<0x4E>(x));
  x = fmaxf(x, dpp_xf<0x141>(x));
  x = fmaxf(x, dpp_xf<0x140>(x));
  return x;
}
__device__ __forceinline__ float red16_sum(float x) {
  x += dpp_xf<0xB1>(x);
  x += dpp_xf<0x4E>(x);
  x += dpp_xf<0x141>(x);
  x += dpp_xf<0x140>(x);
  return x;
}

// A-matrix 16x32 bf16 fragment (ISA 7.12.2): lane l -> row m = l&15, half = l>>4.
// element i: k = i + (i>=8 ? 8 : 0) + half*8  => two contiguous 16B chunks at
// k-offsets half*8 and 16+half*8. `base` points at [row0][k0]; stride in elems.
__device__ __forceinline__ v16bf load_frag_A(const unsigned short* base, int stride) {
  int l = threadIdx.x & 31;
  int m = l & 15, half = l >> 4;
  const unsigned short* p = base + (size_t)m * stride;
  Frag16 f;
  f.u[0] = *(const uint4*)(p + half * 8);
  f.u[1] = *(const uint4*)(p + 16 + half * 8);
  return f.v;
}

// B-matrix 32x16 bf16 fragment: lane l -> col n = l&15, half = l>>4.
// element i: k = i + (i>=8 ? 8 : 0) + half*8 (same as A). Source is stored
// "Bt" style: row n holds the K-contiguous data (stride in elems).
__device__ __forceinline__ v16bf load_frag_B(const unsigned short* base, int stride) {
  int l = threadIdx.x & 31;
  int n = l & 15, half = l >> 4;
  const unsigned short* p = base + (size_t)n * stride;
  Frag16 f;
  f.u[0] = *(const uint4*)(p + half * 8);
  f.u[1] = *(const uint4*)(p + 16 + half * 8);
  return f.v;
}

// ---------------------------------------------------------------------------
// Problem constants (from the reference)
// ---------------------------------------------------------------------------
#define BB   2
#define TT   2048
#define CC   1024
#define HH   16
#define DD   64
#define MM   (BB * TT)        // 4096 tokens

// ---------------------------------------------------------------------------
// Kernel 1: fp32 -> bf16 convert
// ---------------------------------------------------------------------------
__global__ void cvt_f32_bf16(const float* __restrict__ in, unsigned short* __restrict__ out, int n) {
  int i = blockIdx.x * blockDim.x + threadIdx.x;
  if (i < n) out[i] = f2bf(in[i]);
}

// Kernel 2: transpose fp32 [K][N] -> bf16 [N][K]
__global__ void transpose_w(const float* __restrict__ W, unsigned short* __restrict__ Wt,
                            int K, int N) {
  int i = blockIdx.x * blockDim.x + threadIdx.x;
  if (i < K * N) {
    int k = i / N, n = i % N;
    Wt[(size_t)n * K + k] = f2bf(W[(size_t)k * N + n]);
  }
}

// ---------------------------------------------------------------------------
// Kernel 3: bf16 WMMA GEMM:  out[M,N] = A[M,K] * Bt[N,K]^T + bias
// Wave tile 32x64 (2 A-frags x 4 B-frags = 8 WMMAs / k-step).
// Ping-pong software pipeline: two named fragment sets alternate (no register
// rotation copies); loads for step s+1 issue before the WMMAs for step s.
// __launch_bounds__(256,1): full VGPR budget, no spills.
// Block = 256 thr = 8 waves (2 in M x 4 in N) => block tile 64 x 256.
// MODE 0: QKV epilogue -> scatter bf16 q (scaled 1/sqrt(d)) [b,h,t,d],
//         k [b,h,t,d], v transposed [b,h,d,t]
// MODE 1: fp32 store to outF[M,N]
// ---------------------------------------------------------------------------
struct FragSet { v16bf a0, a1, b0, b1, b2, b3; };

__device__ __forceinline__ void load_set(FragSet& fs,
                                         const unsigned short* Arow0,
                                         const unsigned short* Arow1,
                                         const unsigned short* Bt,
                                         size_t n0, int k0, int K) {
  fs.a0 = load_frag_A(Arow0 + k0, K);
  fs.a1 = load_frag_A(Arow1 + k0, K);
  fs.b0 = load_frag_B(Bt + (n0 +  0) * K + k0, K);
  fs.b1 = load_frag_B(Bt + (n0 + 16) * K + k0, K);
  fs.b2 = load_frag_B(Bt + (n0 + 32) * K + k0, K);
  fs.b3 = load_frag_B(Bt + (n0 + 48) * K + k0, K);
}

__device__ __forceinline__ void mma_set(const FragSet& fs, v8f (&acc)[2][4]) {
  acc[0][0] = WMMA_BF16(fs.a0, fs.b0, acc[0][0]);
  acc[1][0] = WMMA_BF16(fs.a1, fs.b0, acc[1][0]);
  acc[0][1] = WMMA_BF16(fs.a0, fs.b1, acc[0][1]);
  acc[1][1] = WMMA_BF16(fs.a1, fs.b1, acc[1][1]);
  acc[0][2] = WMMA_BF16(fs.a0, fs.b2, acc[0][2]);
  acc[1][2] = WMMA_BF16(fs.a1, fs.b2, acc[1][2]);
  acc[0][3] = WMMA_BF16(fs.a0, fs.b3, acc[0][3]);
  acc[1][3] = WMMA_BF16(fs.a1, fs.b3, acc[1][3]);
}

template <int K, int MODE>
__global__ void __launch_bounds__(256, 1)
gemm_bf16(const unsigned short* __restrict__ A,
          const unsigned short* __restrict__ Bt,
          const float* __restrict__ bias,
          int M, int N,
          float* __restrict__ outF,
          unsigned short* __restrict__ qbf,
          unsigned short* __restrict__ kbf,
          unsigned short* __restrict__ vTbf) {
  int wave = threadIdx.x >> 5;
  int wm = wave >> 2;               // 0..1
  int wn = wave & 3;                // 0..3
  int m0 = blockIdx.x * 64 + wm * 32;
  int n0 = blockIdx.y * 256 + wn * 64;
  if (m0 >= M || n0 >= N) return;

  const unsigned short* Arow0 = A + (size_t)m0 * K;
  const unsigned short* Arow1 = A + (size_t)(m0 + 16) * K;
  int lane = threadIdx.x & 31;

  v8f acc[2][4] = {};
  FragSet f0, f1;

  constexpr int NSTEP = K / 32;     // 32 for K=1024 (even)
  load_set(f0, Arow0, Arow1, Bt, (size_t)n0, 0, K);

  #pragma unroll 1
  for (int s = 0; s + 2 <= NSTEP - 1; s += 2) {
    // prefetch 2 steps ahead into near caches (global_prefetch_b8)
    __builtin_prefetch(Bt + (size_t)(n0 + lane) * K + (s + 2) * 32, 0, 3);
    load_set(f1, Arow0, Arow1, Bt, (size_t)n0, (s + 1) * 32, K);
    mma_set(f0, acc);                       // step s
    load_set(f0, Arow0, Arow1, Bt, (size_t)n0, (s + 2) * 32, K);
    mma_set(f1, acc);                       // step s+1
  }
  // tail: steps NSTEP-2 (in f0) and NSTEP-1
  load_set(f1, Arow0, Arow1, Bt, (size_t)n0, (NSTEP - 1) * 32, K);
  mma_set(f0, acc);
  mma_set(f1, acc);

  int nlane = lane & 15, half = lane >> 4;
  #pragma unroll
  for (int g = 0; g < 2; ++g) {
    #pragma unroll
    for (int t = 0; t < 4; ++t) {
      int n = n0 + t * 16 + nlane;
      float bv = bias[n];
      #pragma unroll
      for (int r = 0; r < 8; ++r) {
        int m = m0 + g * 16 + r + half * 8;
        float v = acc[g][t][r] + bv;
        if (MODE == 1) {
          outF[(size_t)m * N + n] = v;
        } else {
          int sec = n >> 10;            // 0=q 1=k 2=v
          int c   = n & 1023;
          int h   = c >> 6;
          int dcol = c & 63;
          int b   = m >> 11;            // token -> batch
          int tok = m & 2047;
          size_t bh = (size_t)b * HH + h;
          if (sec == 0) {
            qbf[(bh * TT + tok) * DD + dcol] = f2bf(v * 0.125f);   // 1/sqrt(64)
          } else if (sec == 1) {
            kbf[(bh * TT + tok) * DD + dcol] = f2bf(v);
          } else {
            vTbf[(bh * DD + dcol) * TT + tok] = f2bf(v);           // transposed
          }
        }
      }
    }
  }
}

// ---------------------------------------------------------------------------
// Kernel 4: flash attention (prefix-causal). One wave = one 16-row query tile.
// 8 waves/block => block covers 128 consecutive query rows of one (b,h).
// Two-phase key loop: bulk blocks (fully visible under causal+prefix mask)
// run with NO mask compares/selects; only boundary blocks take the masked
// path. Row softmax reductions use DPP (VALU) instead of LDS bpermute.
// K-fragments reload in place (no rotation copies); V loads + next-K loads
// issue before the softmax VALU so latency is hidden.
// __launch_bounds__(256,1): full VGPR budget, no spills.
// P staged through per-wave LDS with explicit s_wait_dscnt (no barrier, so
// waves may run different causal trip counts).
// ---------------------------------------------------------------------------
__global__ void __launch_bounds__(256, 1)
attn_flash(const unsigned short* __restrict__ qbf,
           const unsigned short* __restrict__ kbf,
           const unsigned short* __restrict__ vTbf,
           unsigned short* __restrict__ ybf,
           const int* __restrict__ Sp) {
  __shared__ unsigned short Plds[8][16 * 32];   // 1KB per wave
  const int S = Sp[0];

  int wave  = threadIdx.x >> 5;
  int l     = threadIdx.x & 31;
  int nlane = l & 15, half = l >> 4;

  int chunk = blockIdx.x & 15;        // T/128 = 16 chunks
  int bh    = blockIdx.x >> 4;        // 0..B*H-1
  int b     = bh >> 4;
  int h     = bh & 15;
  int qt0   = chunk * 128 + wave * 16;

  const unsigned short* qh = qbf + (size_t)bh * TT * DD;
  const unsigned short* kh = kbf + (size_t)bh * TT * DD;
  const unsigned short* vh = vTbf + (size_t)bh * DD * TT;
  unsigned short* Pw = &Plds[wave][0];

  v8f o[4] = {};
  float mrun[8], lrun[8];
  int  irow[8];
  bool pre[8];
  #pragma unroll
  for (int r = 0; r < 8; ++r) {
    mrun[r] = -3.0e38f; lrun[r] = 0.0f;
    irow[r] = qt0 + r + half * 8;      // loop-invariant query row index
    pre[r]  = irow[r] < S;             // loop-invariant prefix flag
  }

  // Q fragments: head-dim chunks [0,32) and [32,64)
  v16bf qf0 = load_frag_A(qh + (size_t)qt0 * DD + 0, DD);
  v16bf qf1 = load_frag_A(qh + (size_t)qt0 * DD + 32, DD);

  int kend = (qt0 + 16 > S) ? (qt0 + 16) : S;   // keys needed: [0, kend)
  int nkb  = (kend + 31) >> 5;

  // number of fully-visible (unmasked) leading blocks
  int nfull1 = (qt0 >= 31) ? (((qt0 - 31) >> 5) + 1) : 0;   // causal-full
  int nfull2 = (qt0 + 15 < S) ? (S >> 5) : 0;               // prefix-full
  int nfull  = nfull1 > nfull2 ? nfull1 : nfull2;
  if (nfull > nkb) nfull = nkb;

  // prologue: K fragments for kb = 0
  v16bf kc00 = load_frag_B(kh + 0, DD);
  v16bf kc01 = load_frag_B(kh + 32, DD);
  v16bf kc10 = load_frag_B(kh + (size_t)16 * DD + 0, DD);
  v16bf kc11 = load_frag_B(kh + (size_t)16 * DD + 32, DD);

  // shared per-block body; `masked` constant-folds after inlining
  auto body = [&](int kb, bool masked) __attribute__((always_inline)) {
    int k0 = kb * 32;

    // scores for current key block (kc* dead after these)
    v8f s0 = {}, s1 = {};
    s0 = WMMA_BF16(qf0, kc00, s0);
    s0 = WMMA_BF16(qf1, kc01, s0);
    s1 = WMMA_BF16(qf0, kc10, s1);
    s1 = WMMA_BF16(qf1, kc11, s1);

    // V fragments for current block — issued early, consumed after softmax
    v16bf vf0 = load_frag_B(vh + (size_t) 0 * TT + k0, TT);
    v16bf vf1 = load_frag_B(vh + (size_t)16 * TT + k0, TT);
    v16bf vf2 = load_frag_B(vh + (size_t)32 * TT + k0, TT);
    v16bf vf3 = load_frag_B(vh + (size_t)48 * TT + k0, TT);

    // next-block K fragments straight into kc* (clamped on last iter)
    int knx = (kb + 1 < nkb) ? k0 + 32 : k0;
    kc00 = load_frag_B(kh + (size_t)knx * DD + 0, DD);
    kc01 = load_frag_B(kh + (size_t)knx * DD + 32, DD);
    kc10 = load_frag_B(kh + (size_t)(knx + 16) * DD + 0, DD);
    kc11 = load_frag_B(kh + (size_t)(knx + 16) * DD + 32, DD);

    // mask only on boundary blocks: visible iff (j <= i) || (i < S && j < S)
    float p0[8], p1[8];
    if (masked) {
      int j0 = k0 + nlane, j1 = k0 + 16 + nlane;
      bool j0pre = j0 < S, j1pre = j1 < S;
      #pragma unroll
      for (int r = 0; r < 8; ++r) {
        p0[r] = ((j0 <= irow[r]) || (pre[r] && j0pre)) ? s0[r] : -3.0e38f;
        p1[r] = ((j1 <= irow[r]) || (pre[r] && j1pre)) ? s1[r] : -3.0e38f;
      }
    } else {
      #pragma unroll
      for (int r = 0; r < 8; ++r) { p0[r] = s0[r]; p1[r] = s1[r]; }
    }

    // online softmax update; row reductions via DPP (no LDS pipe)
    #pragma unroll
    for (int r = 0; r < 8; ++r) {
      float mx = red16_max(fmaxf(p0[r], p1[r]));
      float mnew = fmaxf(mrun[r], mx);
      float e0 = __expf(p0[r] - mnew);
      float e1 = __expf(p1[r] - mnew);
      float rs = red16_sum(e0 + e1);
      float scale = __expf(mrun[r] - mnew);
      lrun[r] = lrun[r] * scale + rs;
      mrun[r] = mnew;
      o[0][r] *= scale; o[1][r] *= scale; o[2][r] *= scale; o[3][r] *= scale;
      int row = r + half * 8;
      Pw[row * 32 + nlane]      = f2bf(e0);
      Pw[row * 32 + 16 + nlane] = f2bf(e1);
    }

    // drain LDS stores before cross-lane fragment reads (same wave, lockstep)
    asm volatile("s_wait_dscnt 0" ::: "memory");

    v16bf pa = load_frag_A(Pw, 32);          // P tile 16x32 as A fragment
    o[0] = WMMA_BF16(pa, vf0, o[0]);
    o[1] = WMMA_BF16(pa, vf1, o[1]);
    o[2] = WMMA_BF16(pa, vf2, o[2]);
    o[3] = WMMA_BF16(pa, vf3, o[3]);
  };

  int kb = 0;
  #pragma unroll 1
  for (; kb < nfull; ++kb) body(kb, false);   // bulk: no mask math
  #pragma unroll 1
  for (; kb < nkb; ++kb) body(kb, true);      // boundary: masked

  // normalize and store y as bf16 [b, t, h*64 + d]
  #pragma unroll
  for (int r = 0; r < 8; ++r) {
    float inv = 1.0f / lrun[r];
    int tok = qt0 + r + half * 8;
    size_t rowbase = ((size_t)b * TT + tok) * CC + h * DD;
    #pragma unroll
    for (int t = 0; t < 4; ++t) {
      ybf[rowbase + t * 16 + nlane] = f2bf(o[t][r] * inv);
    }
  }
}

// ---------------------------------------------------------------------------
// Host side
// ---------------------------------------------------------------------------
extern "C" void kernel_launch(void* const* d_in, const int* in_sizes, int n_in,
                              void* d_out, int out_size, void* d_ws, size_t ws_size,
                              hipStream_t stream) {
  const float* x     = (const float*)d_in[0];   // [B,T,C]
  const float* Wqkv  = (const float*)d_in[1];   // [C,3C]
  const float* bqkv  = (const float*)d_in[2];   // [3C]
  const float* Wproj = (const float*)d_in[3];   // [C,C]
  const float* bproj = (const float*)d_in[4];   // [C]
  const int*   Sp    = (const int*)d_in[5];     // scalar S
  float*       out   = (float*)d_out;           // [B,T,C]

  unsigned short* ws = (unsigned short*)d_ws;
  size_t off = 0;
  unsigned short* xbf    = ws + off; off += (size_t)MM * CC;        // 4096x1024
  unsigned short* WqkvT  = ws + off; off += (size_t)(3 * CC) * CC;  // 3072x1024
  unsigned short* WprojT = ws + off; off += (size_t)CC * CC;        // 1024x1024
  unsigned short* qbf    = ws + off; off += (size_t)BB * HH * TT * DD;
  unsigned short* kbf    = ws + off; off += (size_t)BB * HH * TT * DD;
  unsigned short* vTbf   = ws + off; off += (size_t)BB * HH * DD * TT;
  unsigned short* ybf    = ws + off; off += (size_t)MM * CC;
  // total: ~48 MiB of bf16 scratch (fits easily; also fits in 192MB L2)

  // 1) convert x to bf16
  {
    int n = MM * CC;
    cvt_f32_bf16<<<(n + 255) / 256, 256, 0, stream>>>(x, xbf, n);
  }
  // 2) transpose weights to bf16 [N][K]
  {
    int n = CC * 3 * CC;
    transpose_w<<<(n + 255) / 256, 256, 0, stream>>>(Wqkv, WqkvT, CC, 3 * CC);
  }
  {
    int n = CC * CC;
    transpose_w<<<(n + 255) / 256, 256, 0, stream>>>(Wproj, WprojT, CC, CC);
  }
  // 3) QKV GEMM: [4096,1024] x [1024,3072] + bias -> scattered q/k/vT bf16
  {
    dim3 grid(MM / 64, (3 * CC) / 256);
    gemm_bf16<CC, 0><<<grid, 256, 0, stream>>>(xbf, WqkvT, bqkv, MM, 3 * CC,
                                               nullptr, qbf, kbf, vTbf);
  }
  // 4) flash attention: 512 blocks x 8 waves
  {
    dim3 grid(BB * HH * (TT / 128));
    attn_flash<<<grid, 256, 0, stream>>>(qbf, kbf, vTbf, ybf, Sp);
  }
  // 5) output projection: [4096,1024] x [1024,1024] + bias -> fp32 out
  {
    dim3 grid(MM / 64, CC / 256);
    gemm_bf16<CC, 1><<<grid, 256, 0, stream>>>(ybf, WprojT, bproj, MM, CC,
                                               out, nullptr, nullptr, nullptr);
  }
}
